// LSTM_Stack_59081570125201
// MI455X (gfx1250) — matrix operand order repaired and checked by reading, not compiled, not run on hardware
//
#include <hip/hip_runtime.h>

// ---------------- constants ----------------
#define INPUT_DIM  128
#define HIDDEN_DIM 512
#define STACK_DIM  64
#define BATCH      128
#define MAX_T      259
#define SEQ_L      259
#define GDIM       (4*HIDDEN_DIM)          // 2048
#define KCAT       (INPUT_DIM+STACK_DIM+HIDDEN_DIM) // 704
#define NR_PAD     208                     // 194 padded to 13*16
#define NR_REAL    (INPUT_DIM+STACK_DIM+2) // 194
#define KSPLIT     128                     // k = (259-3)/2

typedef __bf16 v16bf __attribute__((ext_vector_type(16)));
typedef float  v8f   __attribute__((ext_vector_type(8)));

union Frag { v16bf v; uint4 q[2]; };

__device__ __forceinline__ unsigned short f2b(float f) {
  unsigned int u = __float_as_uint(f);
  unsigned int r = u + 0x7FFFu + ((u >> 16) & 1u);
  return (unsigned short)(r >> 16);
}
__device__ __forceinline__ float sigf(float x) { return 1.0f / (1.0f + expf(-x)); }

// ---------------- weight conversion ----------------
__global__ void conv_wcat(const float* __restrict__ Wih, const float* __restrict__ Whh,
                          unsigned short* __restrict__ Wcat) {
  int idx = blockIdx.x * blockDim.x + threadIdx.x;
  if (idx >= GDIM * KCAT) return;
  int r = idx / KCAT, c = idx % KCAT;
  float v = (c < INPUT_DIM + STACK_DIM) ? Wih[r * (INPUT_DIM + STACK_DIM) + c]
                                        : Whh[r * HIDDEN_DIM + (c - (INPUT_DIM + STACK_DIM))];
  Wcat[idx] = f2b(v);
}
__global__ void conv_biasg(const float* __restrict__ bih, const float* __restrict__ bhh,
                           float* __restrict__ bg) {
  int i = blockIdx.x * blockDim.x + threadIdx.x;
  if (i < GDIM) bg[i] = bih[i] + bhh[i];
}
__global__ void conv_wr(const float* __restrict__ Wr, unsigned short* __restrict__ Wrp) {
  int idx = blockIdx.x * blockDim.x + threadIdx.x;
  if (idx >= NR_PAD * HIDDEN_DIM) return;
  int r = idx / HIDDEN_DIM, c = idx % HIDDEN_DIM;
  Wrp[idx] = f2b(r < NR_REAL ? Wr[r * HIDDEN_DIM + c] : 0.0f);
}
__global__ void conv_br(const float* __restrict__ br, float* __restrict__ brp) {
  int i = blockIdx.x * blockDim.x + threadIdx.x;
  if (i < NR_PAD) brp[i] = (i < NR_REAL) ? br[i] : 0.0f;
}
__global__ void zero_u32(unsigned int* p, int n) {
  int i = blockIdx.x * blockDim.x + threadIdx.x;
  if (i < n) p[i] = 0u;
}

// ---------------- per-step kernels ----------------
// act row layout (bf16, stride KCAT=704): [ x_t (128) | r_{t-1} (64) | h_{t-1} (512) ]
__global__ void prep_act(const float* __restrict__ x, unsigned short* __restrict__ act, int t) {
  int idx = blockIdx.x * blockDim.x + threadIdx.x;
  if (idx >= BATCH * INPUT_DIM) return;
  int b = idx / INPUT_DIM, j = idx % INPUT_DIM;
  float v = (t <= KSPLIT) ? x[((size_t)t * BATCH + b) * INPUT_DIM + j] : 0.0f;
  act[(size_t)b * KCAT + j] = f2b(v);
}

// g[B,2048] = act @ Wcat^T + (bih+bhh).  grid=(8,32), block=128 (4 waves, 16 cols each)
__global__ void gates_gemm(const unsigned short* __restrict__ act,
                           const unsigned short* __restrict__ W,
                           const float* __restrict__ bias,
                           float* __restrict__ g) {
  const int lane = threadIdx.x & 31;
  const int wv   = threadIdx.x >> 5;
  const int l16  = lane & 15;
  const int hi   = lane >> 4;
  const int m0 = blockIdx.x * 16;
  const int n0 = blockIdx.y * 64 + wv * 16;
  const int m = m0 + l16;
  const int n = n0 + l16;

  const uint4* aq = reinterpret_cast<const uint4*>(act) + (size_t)m * (KCAT / 8) + hi;
  const uint4* wq = reinterpret_cast<const uint4*>(W)   + (size_t)n * (KCAT / 8) + hi;

  v8f acc = {};
#pragma unroll
  for (int kb = 0; kb < KCAT / 32; ++kb) {
    Frag A, B;
    A.q[0] = aq[4 * kb];        // K = k0 + hi*8 + 0..7
    A.q[1] = aq[4 * kb + 2];    // K = k0 + hi*8 + 16..23
    B.q[0] = wq[4 * kb];        // K = k0 + hi*8 + 0..7   (row n of W == col n of W^T)
    B.q[1] = wq[4 * kb + 2];    // K = k0 + hi*8 + 16..23
    __builtin_prefetch(aq + 4 * kb + 8, 0, 3);
    acc = __builtin_amdgcn_wmma_f32_16x16x32_bf16(false, A.v, false, B.v,
                                                  (short)0, acc, false, false);
  }
  const float bcol = bias[n0 + l16];
  float* grow = g + (size_t)(m0 + hi * 8) * GDIM + n0 + l16;
#pragma unroll
  for (int r = 0; r < 8; ++r) grow[(size_t)r * GDIM] = acc[r] + bcol;
}

// c,h update; h written back as bf16 into act[.,192:704]
__global__ void lstm_cell(const float* __restrict__ g, float* __restrict__ c,
                          unsigned short* __restrict__ act) {
  int idx = blockIdx.x * blockDim.x + threadIdx.x;
  if (idx >= BATCH * HIDDEN_DIM) return;
  int b = idx / HIDDEN_DIM, j = idx % HIDDEN_DIM;
  const float* gr = g + (size_t)b * GDIM;
  float gi = gr[j], gf = gr[j + 512], gg = gr[j + 1024], go = gr[j + 1536];
  float c2 = sigf(gf) * c[idx] + sigf(gi) * tanhf(gg);
  float h2 = sigf(go) * tanhf(c2);
  c[idx] = c2;
  act[(size_t)b * KCAT + (INPUT_DIM + STACK_DIM) + j] = f2b(h2);
}

// out[B,208] = h @ Wr^T + br.  grid=(8,13), block=32 (one wave)
__global__ void out_gemm(const unsigned short* __restrict__ act,
                         const unsigned short* __restrict__ Wrp,
                         const float* __restrict__ brp,
                         float* __restrict__ out) {
  const int lane = threadIdx.x & 31;
  const int l16  = lane & 15;
  const int hi   = lane >> 4;
  const int m0 = blockIdx.x * 16;
  const int n0 = blockIdx.y * 16;
  const int m = m0 + l16;
  const int n = n0 + l16;

  const uint4* aq = reinterpret_cast<const uint4*>(act) + (size_t)m * (KCAT / 8)
                    + (INPUT_DIM + STACK_DIM) / 8 + hi;
  const uint4* wq = reinterpret_cast<const uint4*>(Wrp) + (size_t)n * (HIDDEN_DIM / 8) + hi;

  v8f acc = {};
#pragma unroll
  for (int kb = 0; kb < HIDDEN_DIM / 32; ++kb) {
    Frag A, B;
    A.q[0] = aq[4 * kb];
    A.q[1] = aq[4 * kb + 2];
    B.q[0] = wq[4 * kb];
    B.q[1] = wq[4 * kb + 2];
    acc = __builtin_amdgcn_wmma_f32_16x16x32_bf16(false, A.v, false, B.v,
                                                  (short)0, acc, false, false);
  }
  const float bcol = brp[n0 + l16];
  float* orow = out + (size_t)(m0 + hi * 8) * NR_PAD + n0 + l16;
#pragma unroll
  for (int r = 0; r < 8; ++r) orow[(size_t)r * NR_PAD] = acc[r] + bcol;
}

// neural-stack update: one block (64 threads) per batch row
__global__ void stack_update(const float* __restrict__ out, float* __restrict__ V,
                             float* __restrict__ s, unsigned short* __restrict__ act, int t) {
  __shared__ float prod[MAX_T];
  __shared__ float Aa[MAX_T];
  const int b = blockIdx.x;
  const int tid = threadIdx.x;
  const float* ob = out + (size_t)b * NR_PAD;
  const float u    = sigf(ob[STACK_DIM + INPUT_DIM]);      // col 192
  const float dval = sigf(ob[STACK_DIM + INPUT_DIM + 1]);  // col 193
  float* sb = s + (size_t)b * MAX_T;

  if (tid == 0) {  // suffix sums of OLD s: prod[j] = sum_{i>j} s[i]
    float run = 0.0f;
    for (int j = t; j >= 0; --j) { prod[j] = run; run += sb[j]; }
  }
  __syncthreads();
  for (int j = tid; j <= t; j += 64) {
    float sj = sb[j];
    float sp = fmaxf(0.0f, sj - fmaxf(0.0f, u - prod[j]));
    if (j == t) sp = dval;
    float inner = fmaxf(0.0f, 1.0f - prod[j] - sp);
    Aa[j] = fminf(sp, inner);
    sb[j] = sp;
  }
  __syncthreads();
  const int d = tid;  // 0..63
  float* Vb = V + ((size_t)b * MAX_T) * STACK_DIM + d;
  float vnew = Vb[(size_t)t * STACK_DIM] + ob[d];  // V[:,t,:] += vt
  Vb[(size_t)t * STACK_DIM] = vnew;
  float r = 0.0f;
  for (int j = 0; j < t; ++j) r += Aa[j] * Vb[(size_t)j * STACK_DIM];
  r += Aa[t] * vnew;
  act[(size_t)b * KCAT + INPUT_DIM + d] = f2b(r);  // r_t for next step
}

// log_softmax over ot = out[:,64:192]; one block (128 threads) per batch row
__global__ void write_logsoftmax(const float* __restrict__ out, float* __restrict__ dst, int to) {
  __shared__ float red[INPUT_DIM];
  const int b = blockIdx.x, tid = threadIdx.x;
  float xv = out[(size_t)b * NR_PAD + STACK_DIM + tid];
  red[tid] = xv;
  __syncthreads();
  for (int sft = 64; sft > 0; sft >>= 1) {
    if (tid < sft) red[tid] = fmaxf(red[tid], red[tid + sft]);
    __syncthreads();
  }
  float mx = red[0];
  __syncthreads();
  red[tid] = expf(xv - mx);
  __syncthreads();
  for (int sft = 64; sft > 0; sft >>= 1) {
    if (tid < sft) red[tid] += red[tid + sft];
    __syncthreads();
  }
  float lse = mx + logf(red[0]);
  dst[((size_t)to * BATCH + b) * INPUT_DIM + tid] = xv - lse;
}

// ---------------- host ----------------
extern "C" void kernel_launch(void* const* d_in, const int* in_sizes, int n_in,
                              void* d_out, int out_size, void* d_ws, size_t ws_size,
                              hipStream_t stream) {
  (void)in_sizes; (void)n_in; (void)out_size; (void)ws_size;
  const float* x   = (const float*)d_in[0];
  const float* Wih = (const float*)d_in[1];
  const float* bih = (const float*)d_in[2];
  const float* Whh = (const float*)d_in[3];
  const float* bhh = (const float*)d_in[4];
  const float* Wr  = (const float*)d_in[5];
  const float* br  = (const float*)d_in[6];
  float* outp = (float*)d_out;

  // workspace carve-up (256B aligned)
  char* ws = (char*)d_ws;
  size_t off = 0;
  auto carve = [&](size_t bytes) { char* p = ws + off; off = (off + bytes + 255) & ~(size_t)255; return p; };
  unsigned short* Wcat = (unsigned short*)carve((size_t)GDIM * KCAT * 2);
  unsigned short* Wrp  = (unsigned short*)carve((size_t)NR_PAD * HIDDEN_DIM * 2);
  float*          bg   = (float*)carve((size_t)GDIM * 4);
  float*          brp  = (float*)carve((size_t)NR_PAD * 4);
  unsigned short* act  = (unsigned short*)carve((size_t)BATCH * KCAT * 2);
  float*          g    = (float*)carve((size_t)BATCH * GDIM * 4);
  float*          c    = (float*)carve((size_t)BATCH * HIDDEN_DIM * 4);
  float*          outb = (float*)carve((size_t)BATCH * NR_PAD * 4);
  float*          V    = (float*)carve((size_t)BATCH * MAX_T * STACK_DIM * 4);
  float*          s    = (float*)carve((size_t)BATCH * MAX_T * 4);

  // one-time conversions + state zeroing (ws is poisoned before timing)
  conv_wcat <<<(GDIM * KCAT + 255) / 256, 256, 0, stream>>>(Wih, Whh, Wcat);
  conv_biasg<<<(GDIM + 255) / 256, 256, 0, stream>>>(bih, bhh, bg);
  conv_wr   <<<(NR_PAD * HIDDEN_DIM + 255) / 256, 256, 0, stream>>>(Wr, Wrp);
  conv_br   <<<1, 256, 0, stream>>>(br, brp);
  {
    int nAct = BATCH * KCAT / 2;                 // ushort pairs as u32
    int nC   = BATCH * HIDDEN_DIM;
    int nV   = BATCH * MAX_T * STACK_DIM;
    int nS   = BATCH * MAX_T;
    zero_u32<<<(nAct + 255) / 256, 256, 0, stream>>>((unsigned int*)act, nAct);
    zero_u32<<<(nC   + 255) / 256, 256, 0, stream>>>((unsigned int*)c,   nC);
    zero_u32<<<(nV   + 255) / 256, 256, 0, stream>>>((unsigned int*)V,   nV);
    zero_u32<<<(nS   + 255) / 256, 256, 0, stream>>>((unsigned int*)s,   nS);
  }

  for (int t = 0; t < MAX_T; ++t) {
    prep_act  <<<(BATCH * INPUT_DIM + 255) / 256, 256, 0, stream>>>(x, act, t);
    gates_gemm<<<dim3(BATCH / 16, GDIM / 64), 128, 0, stream>>>(act, Wcat, bg, g);
    lstm_cell <<<(BATCH * HIDDEN_DIM + 255) / 256, 256, 0, stream>>>(g, c, act);
    out_gemm  <<<dim3(BATCH / 16, NR_PAD / 16), 32, 0, stream>>>(act, Wrp, brp, outb);
    stack_update<<<BATCH, 64, 0, stream>>>(outb, V, s, act, t);
    if (t >= KSPLIT + 1)
      write_logsoftmax<<<BATCH, INPUT_DIM, 0, stream>>>(outb, outp, t - (KSPLIT + 1));
  }
}
